// IncrementalVGG16_87436944212527
// MI455X (gfx1250) — matrix operand order repaired and checked by reading, not compiled, not run on hardware
//
#include <hip/hip_runtime.h>
#include <hip/hip_bf16.h>

// ---------------------------------------------------------------------------
// VGG16 forward, MI455X (gfx1250, wave32, WMMA f16->f32).
//
//   * activations NHWC f16 ping-pong in d_ws; conv weights pre-transposed to
//     [cout][tap][cin] f16 so every WMMA operand is a contiguous b128 load.
//   * GEMM roles: A = weights (M=cout), B = im2col activations (N=spatial).
//   * Register blocking (new): each wave owns a 64(cout) x 32(spatial) tile:
//       4 A-frags x 2 B-frags -> 8 accumulators, 8 WMMAs per 32-wide k-step
//       from 12 b128 loads (1.5 loads/WMMA vs 4.0 for a 16x16 tile), and
//       8 independent D-chains so the XDL pipe stays fed between loadcnt waits.
//     Wave32 + 1024-VGPR register file makes this cheap (~120 VGPRs live).
//   * Borders: clamped addresses + v_cndmask zeroing, never divergent loads.
//   * FC: M=16 batch GEMM streaming f32 weights (494 MB ~= 21 us @ 23.3 TB/s,
//     pure bandwidth; every weight used once so no blocking needed).
// ---------------------------------------------------------------------------

typedef _Float16 v16h __attribute__((ext_vector_type(16)));
typedef _Float16 v8h  __attribute__((ext_vector_type(8)));
typedef float    v8f  __attribute__((ext_vector_type(8)));

__device__ __forceinline__ v16h cat8(v8h lo, v8h hi) {
  return __builtin_shufflevector(lo, hi, 0,1,2,3,4,5,6,7,8,9,10,11,12,13,14,15);
}

// ---------------- weight transpose: OIHW f32 -> [co][s][ci] f16 ----------------
__global__ void vgg_wtrans(const float* __restrict__ w, _Float16* __restrict__ wt,
                           int Cin, int Cout)
{
  const long i = (long)blockIdx.x * blockDim.x + threadIdx.x;
  const long total = (long)Cout * Cin * 9;
  if (i >= total) return;
  const int  ci   = (int)(i % Cin);
  const long rest = i / Cin;
  const int  s    = (int)(rest % 9);
  const long co   = rest / 9;
  wt[i] = (_Float16)w[(co * Cin + ci) * 9 + s];   // dst idx = co*9*Cin + s*Cin + ci
}

// layer-1 weights: Cin=3, K=27 padded to 32 per cout, order (s, ci)
__global__ void vgg_wtrans1(const float* __restrict__ w, _Float16* __restrict__ wt)
{
  const int i = blockIdx.x * blockDim.x + threadIdx.x;
  if (i >= 64 * 32) return;
  const int k  = i & 31;
  const int co = i >> 5;
  _Float16 v = (_Float16)0.f;
  if (k < 27) {
    const int s  = k / 3;
    const int ci = k - 3 * s;
    v = (_Float16)w[(co * 3 + ci) * 9 + s];
  }
  wt[i] = v;
}

// ---------------- conv layer 1: NCHW f32 in (Cin=3) -> NHWC f16 out ----------------
// grid: (HW/64, Cout/16, batch), block (32,4): threadIdx.y = spatial subtile
__global__ void __launch_bounds__(128)
vgg_conv1(const float* __restrict__ x, const _Float16* __restrict__ wt,
          const float* __restrict__ bias, _Float16* __restrict__ out,
          int Cout, int H, int W)
{
  const int lane = threadIdx.x, half = lane >> 4, row = lane & 15;
  const int HW = H * W;
  const int mbase = blockIdx.x * 64 + threadIdx.y * 16;
  if (mbase >= HW) return;                       // wave-uniform
  const int n = blockIdx.z;
  const int m = mbase + row;
  const bool mvalid = (m < HW);
  const int mc = mvalid ? m : (HW - 1);
  const int y = mc / W, xx = mc - y * W;

  const _Float16* aRow = wt + (long)(blockIdx.y * 16 + row) * 32 + (half ? 8 : 0);
  const v8h a_lo = *(const v8h*)(aRow);
  const v8h a_hi = *(const v8h*)(aRow + 16);

  v16h b;
  #pragma unroll
  for (int t = 0; t < 16; ++t) {
    const int k  = (t < 8) ? half * 8 + t : 16 + half * 8 + (t - 8);  // 0..31
    const int s  = k / 3;
    const int ci = k - 3 * s;
    const int sc = (s < 9) ? s : 8;
    const int ky = sc / 3, kx = sc - 3 * (sc / 3);
    const int iy = y + ky - 1, ix = xx + kx - 1;
    const bool ok = mvalid && (k < 27) &&
                    ((unsigned)iy < (unsigned)H) && ((unsigned)ix < (unsigned)W);
    const int iyc = iy < 0 ? 0 : (iy >= H ? H - 1 : iy);
    const int ixc = ix < 0 ? 0 : (ix >= W ? W - 1 : ix);
    const float fv = x[((long)(n * 3 + ci) * H + iyc) * W + ixc];  // always in-range
    b[t] = ok ? (_Float16)fv : (_Float16)0.f;
  }

  v8f c = {};
  c = __builtin_amdgcn_wmma_f32_16x16x32_f16(false, cat8(a_lo, a_hi),
                                             false, b, (short)0, c, false, false);

  const int cobase = blockIdx.y * 16 + half * 8;
  const float4 b0 = ((const float4*)(bias + cobase))[0];
  const float4 b1 = ((const float4*)(bias + cobase))[1];
  const float bv[8] = {b0.x, b0.y, b0.z, b0.w, b1.x, b1.y, b1.z, b1.w};
  if (mvalid) {
    v8h r;
    #pragma unroll
    for (int v = 0; v < 8; ++v) {
      const float t = c[v] + bv[v];
      r[v] = (_Float16)(t > 0.f ? t : 0.f);
    }
    *(v8h*)(out + ((long)(n * H + y) * W + xx) * Cout + cobase) = r;
  }
}

// ---------------- generic conv: register-tiled 64(cout) x 32(spatial) per wave ----
// grid: (ceil(HW/128), Cout/64, batch), block (32,4): threadIdx.y = spatial chunk
__global__ void __launch_bounds__(128)
vgg_convN(const _Float16* __restrict__ in, const _Float16* __restrict__ wt,
          const float* __restrict__ bias, _Float16* __restrict__ out,
          int Cin, int Cout, int H, int W)
{
  const int lane = threadIdx.x, half = lane >> 4, row = lane & 15;
  const int HW = H * W;
  const int K  = Cin * 9;
  const int sbase = blockIdx.x * 128 + threadIdx.y * 32;   // 32 spatial per wave
  if (sbase >= HW) return;                                 // wave-uniform
  const int n   = blockIdx.z;
  const int coB = blockIdx.y * 64;
  const long nb = (long)n * H;

  // two spatial sub-tiles of 16 (B-fragment columns)
  int  yy[2], xv[2];
  bool mv[2];
  #pragma unroll
  for (int i = 0; i < 2; ++i) {
    const int m = sbase + i * 16 + row;
    mv[i] = (m < HW);
    const int mc = mv[i] ? m : (HW - 1);
    yy[i] = mc / W;  xv[i] = mc - yy[i] * W;
  }

  // four cout sub-tiles of 16 (A-fragment rows); identical addresses across
  // the 4 waves of the block -> weight lines stay hot in WGP$/L2
  const _Float16* aRow[4];
  #pragma unroll
  for (int j = 0; j < 4; ++j)
    aRow[j] = wt + (long)(coB + j * 16 + row) * K + (half ? 8 : 0);

  v8f acc[4][2] = {};
  const v8h z = {};

  for (int s = 0; s < 9; ++s) {                  // 3x3 taps: predicates hoisted
    const int ky = s / 3, kx = s - 3 * (s / 3);
    const _Float16* bptr[2];
    bool ok[2];
    #pragma unroll
    for (int i = 0; i < 2; ++i) {
      const int iy = yy[i] + ky - 1, ix = xv[i] + kx - 1;
      ok[i] = mv[i] && ((unsigned)iy < (unsigned)H) && ((unsigned)ix < (unsigned)W);
      const int iyc = iy < 0 ? 0 : (iy >= H ? H - 1 : iy);
      const int ixc = ix < 0 ? 0 : (ix >= W ? W - 1 : ix);
      bptr[i] = in + ((nb + iyc) * W + ixc) * Cin + (half ? 8 : 0);
    }
    const long aoff = (long)s * Cin;
    for (int c0 = 0; c0 < Cin; c0 += 32) {       // 12 b128 loads -> 8 WMMAs
      v16h bfr[2];
      #pragma unroll
      for (int i = 0; i < 2; ++i) {
        v8h blo = *(const v8h*)(bptr[i] + c0);
        v8h bhi = *(const v8h*)(bptr[i] + c0 + 16);
        blo = ok[i] ? blo : z;                   // v_cndmask, no exec branches
        bhi = ok[i] ? bhi : z;
        bfr[i] = cat8(blo, bhi);
      }
      #pragma unroll
      for (int j = 0; j < 4; ++j) {
        const _Float16* ap = aRow[j] + aoff + c0;
        const v16h afr = cat8(*(const v8h*)ap, *(const v8h*)(ap + 16));
        #pragma unroll
        for (int i = 0; i < 2; ++i)
          acc[j][i] = __builtin_amdgcn_wmma_f32_16x16x32_f16(
              false, afr, false, bfr[i], (short)0, acc[j][i], false, false);
      }
    }
  }

  // epilogue: bias + ReLU, 8x 16B NHWC stores
  #pragma unroll
  for (int j = 0; j < 4; ++j) {
    const int cobase = coB + j * 16 + half * 8;
    const float4 b0 = ((const float4*)(bias + cobase))[0];
    const float4 b1 = ((const float4*)(bias + cobase))[1];
    const float bv[8] = {b0.x, b0.y, b0.z, b0.w, b1.x, b1.y, b1.z, b1.w};
    #pragma unroll
    for (int i = 0; i < 2; ++i) {
      if (mv[i]) {
        v8h r;
        #pragma unroll
        for (int v = 0; v < 8; ++v) {
          const float t = acc[j][i][v] + bv[v];
          r[v] = (_Float16)(t > 0.f ? t : 0.f);
        }
        *(v8h*)(out + ((nb + yy[i]) * W + xv[i]) * Cout + cobase) = r;
      }
    }
  }
}

// ---------------- FC: [16,K] x [K,N] + bias (+ReLU) ----------------
// perm7: FC1 only — our K is NHWC-ordered, reference flatten is NCHW:
//        krow = (k & 511)*49 + (k >> 9)   (exact: C=512=2^9, HW=49)
__global__ void __launch_bounds__(128)
vgg_fc(const _Float16* __restrict__ in, const float* __restrict__ w,
       const float* __restrict__ bias, void* __restrict__ out_,
       int K, int Nout, int ntiles, int do_relu, int out_f32, int perm7)
{
  const int lane = threadIdx.x, half = lane >> 4, row = lane & 15;
  const int ntile = blockIdx.x * 4 + threadIdx.y;
  if (ntile >= ntiles) return;                   // wave-uniform
  const int  co     = ntile * 16 + row;
  const bool nvalid = (co < Nout);
  const int  coc    = nvalid ? co : (Nout - 1);  // clamp -> unconditional loads

  v8f c = {};
  const _Float16* aRow = in + (long)row * K;

  for (int k0 = 0; k0 < K; k0 += 32) {
    const v8h a_lo = *(const v8h*)(aRow + k0 + (half ? 8 : 0));
    const v8h a_hi = *(const v8h*)(aRow + k0 + (half ? 8 : 0) + 16);

    v16h b;
    const int kb = k0 + (half ? 8 : 0);
    #pragma unroll
    for (int t = 0; t < 16; ++t) {
      const int  k    = kb + t + (t < 8 ? 0 : 8);
      const long krow = perm7 ? ((long)(k & 511) * 49 + (k >> 9)) : (long)k;
      const float fv  = w[krow * Nout + coc];    // coalesced across lanes
      b[t] = nvalid ? (_Float16)fv : (_Float16)0.f;
    }
    c = __builtin_amdgcn_wmma_f32_16x16x32_f16(false, cat8(a_lo, a_hi),
                                               false, b, (short)0, c, false, false);
  }

  if (!nvalid) return;
  const float bv = bias[co];
  if (out_f32) {
    float* o = (float*)out_;
    #pragma unroll
    for (int v = 0; v < 8; ++v) {
      float t = c[v] + bv;
      if (do_relu && t < 0.f) t = 0.f;
      o[(long)(half * 8 + v) * Nout + co] = t;
    }
  } else {
    _Float16* o = (_Float16*)out_;
    #pragma unroll
    for (int v = 0; v < 8; ++v) {
      float t = c[v] + bv;
      if (do_relu && t < 0.f) t = 0.f;
      o[(long)(half * 8 + v) * Nout + co] = (_Float16)t;
    }
  }
}

// ---------------- 2x2 maxpool stride 2, NHWC f16, 8 channels/thread ----------------
__global__ void vgg_pool(const _Float16* __restrict__ in, _Float16* __restrict__ out,
                         int C, int H, int W, long total8)
{
  const long i = (long)blockIdx.x * blockDim.x + threadIdx.x;
  if (i >= total8) return;
  const int C8 = C >> 3;
  const int Wo = W >> 1, Ho = H >> 1;
  const int  c8   = (int)(i % C8);
  long       rest = i / C8;
  const int  xo   = (int)(rest % Wo);  rest /= Wo;
  const int  yo   = (int)(rest % Ho);
  const int  n    = (int)(rest / Ho);
  const _Float16* p = in + (((long)n * H + 2 * yo) * W + 2 * xo) * C + c8 * 8;
  const v8h p00 = *(const v8h*)p;
  const v8h p01 = *(const v8h*)(p + C);
  const v8h p10 = *(const v8h*)(p + (long)W * C);
  const v8h p11 = *(const v8h*)(p + (long)W * C + C);
  v8h r;
  #pragma unroll
  for (int t = 0; t < 8; ++t) {
    _Float16 m0 = p00[t] > p01[t] ? p00[t] : p01[t];
    _Float16 m1 = p10[t] > p11[t] ? p10[t] : p11[t];
    r[t] = m0 > m1 ? m0 : m1;
  }
  *(v8h*)(out + (((long)n * Ho + yo) * Wo + xo) * C + c8 * 8) = r;
}

// ---------------------------------------------------------------------------
extern "C" void kernel_launch(void* const* d_in, const int* in_sizes, int n_in,
                              void* d_out, int out_size, void* d_ws, size_t ws_size,
                              hipStream_t stream) {
  (void)in_sizes; (void)n_in; (void)out_size; (void)ws_size;

  const float* x = (const float*)d_in[0];
  const float *W[13], *B[13];
  for (int i = 0; i < 13; ++i) { W[i] = (const float*)d_in[1 + 2 * i];
                                 B[i] = (const float*)d_in[2 + 2 * i]; }
  const float *FW[3], *FB[3];
  for (int j = 0; j < 3; ++j) { FW[j] = (const float*)d_in[27 + 2 * j];
                                FB[j] = (const float*)d_in[28 + 2 * j]; }

  static const int  cin[13]  = {3,64,64,128,128,256,256,256,512,512,512,512,512};
  static const int  cout[13] = {64,64,128,128,256,256,256,512,512,512,512,512,512};
  static const int  hh[13]   = {224,224,112,112,56,56,56,28,28,28,14,14,14};
  static const bool pAf[13]  = {false,true,false,true,false,false,true,
                                false,false,true,false,false,true};

  // ws layout: bufA @0, bufB @112MB (max activation 103MB), wT @224MB (~29.5MB)
  _Float16* bufA = (_Float16*)d_ws;
  _Float16* bufB = (_Float16*)((char*)d_ws + 112ull * 1024 * 1024);
  _Float16* wtb  = (_Float16*)((char*)d_ws + 224ull * 1024 * 1024);

  _Float16* WT[13];
  { size_t off = 0;
    WT[0] = wtb;  off += 64 * 32;
    for (int i = 1; i < 13; ++i) { WT[i] = wtb + off;
                                   off += (size_t)cout[i] * cin[i] * 9; } }

  // ---- weight transposes (tiny; re-done every launch, deterministic) ----
  vgg_wtrans1<<<dim3((64 * 32 + 255) / 256), dim3(256), 0, stream>>>(W[0], WT[0]);
  for (int i = 1; i < 13; ++i) {
    const long total = (long)cout[i] * cin[i] * 9;
    vgg_wtrans<<<dim3((unsigned)((total + 255) / 256)), dim3(256), 0, stream>>>(
        W[i], WT[i], cin[i], cout[i]);
  }

  // ---- conv stack ----
  _Float16* cur = nullptr;
  _Float16* nxt = bufA;
  for (int i = 0; i < 13; ++i) {
    const int H = hh[i], HW = H * H;
    if (i == 0) {
      dim3 grid((HW + 63) / 64, cout[0] / 16, 16);
      vgg_conv1<<<grid, dim3(32, 4), 0, stream>>>(x, WT[0], B[0], nxt,
                                                  cout[0], H, H);
    } else {
      dim3 grid((HW + 127) / 128, cout[i] / 64, 16);
      vgg_convN<<<grid, dim3(32, 4), 0, stream>>>(cur, WT[i], B[i], nxt,
                                                  cin[i], cout[i], H, H);
    }
    cur = nxt; nxt = (cur == bufA) ? bufB : bufA;
    if (pAf[i]) {
      const int Ho = H / 2;
      const long total8 = 16L * cout[i] * Ho * Ho / 8;
      vgg_pool<<<dim3((unsigned)((total8 + 255) / 256)), dim3(256), 0, stream>>>(
          cur, nxt, cout[i], H, H, total8);
      cur = nxt; nxt = (cur == bufA) ? bufB : bufA;
    }
  }

  // ---- FC stack ----
  { const int K = 25088, N = 4096, nt = N / 16;           // FC1 (perm NHWC->NCHW)
    vgg_fc<<<dim3(nt / 4), dim3(32, 4), 0, stream>>>(cur, FW[0], FB[0], nxt,
                                                     K, N, nt, 1, 0, 1);
    cur = nxt; nxt = (cur == bufA) ? bufB : bufA; }
  { const int K = 4096, N = 4096, nt = N / 16;            // FC2
    vgg_fc<<<dim3(nt / 4), dim3(32, 4), 0, stream>>>(cur, FW[1], FB[1], nxt,
                                                     K, N, nt, 1, 0, 0);
    cur = nxt; nxt = (cur == bufA) ? bufB : bufA; }
  { const int K = 4096, N = 1000, nt = (N + 15) / 16;     // FC3 -> f32 d_out
    vgg_fc<<<dim3((nt + 3) / 4), dim3(32, 4), 0, stream>>>(cur, FW[2], FB[2], d_out,
                                                           K, N, nt, 0, 1, 0); }
}
